// GATLayer_22617297780843
// MI455X (gfx1250) — matrix operand order repaired and checked by reading, not compiled, not run on hardware
//
#include <hip/hip_runtime.h>
#include <hip/hip_bf16.h>

// ---- problem constants (from reference) ----
#define NNODES 50000
#define NEDGES 800000
#define IN_F   128
#define NRELA  64
#define OUT_F  128
#define KDIM   320            // 2*IN_F + NRELA
#define KTILES 10             // 320 / 32
#define ETILE  64             // edges per block
#define A_STRIDE 328          // padded LDS row stride (halves)
#define LRELU  0.2f

// ---- workspace layout (bytes) ----
#define WS_WT     0                       // bf16 W^T [OUT_F][KDIM]  : 81920 B
#define WS_ALPHA  81920                   // f32 per-edge alpha / e  : 3.2 MB
#define WS_S      3281920                 // f32 per-node sum        : 200 KB
#define WS_M      3481920                 // u32 ordered per-node max: 200 KB
// total ~3.68 MB

typedef __attribute__((ext_vector_type(16))) __bf16 v16bf;
typedef __attribute__((ext_vector_type(8)))  __bf16 v8bf;
typedef __attribute__((ext_vector_type(8)))  float  v8f;

__device__ __forceinline__ unsigned short f2bf_u16(float f) {
    unsigned u = __float_as_uint(f);
    unsigned r = (u + 0x7FFFu + ((u >> 16) & 1u)) >> 16;   // RNE
    return (unsigned short)r;
}
// order-preserving float <-> uint mapping for atomicMax on floats
__device__ __forceinline__ unsigned f2ord(float f) {
    unsigned u = __float_as_uint(f);
    return (int)u >= 0 ? (u | 0x80000000u) : ~u;
}
__device__ __forceinline__ float ord2f(unsigned u) {
    return (u & 0x80000000u) ? __uint_as_float(u & 0x7FFFFFFFu)
                             : __uint_as_float(~u);
}

// ---------------------------------------------------------------------------
// Kernel 0: convert W -> bf16 W^T in ws, init per-node max/sum, zero out.
// ---------------------------------------------------------------------------
__global__ void __launch_bounds__(256)
gat_prep(const float* __restrict__ W, unsigned short* __restrict__ Wt,
         float* __restrict__ s, unsigned* __restrict__ m_ord,
         float* __restrict__ out) {
    int i = blockIdx.x * 256 + threadIdx.x;
    if (i < KDIM * OUT_F) {             // Wt[n][k] = W[k][n]
        int n = i / KDIM, k = i % KDIM;
        Wt[i] = f2bf_u16(W[k * OUT_F + n]);
    }
    if (i < NNODES) { s[i] = 0.f; m_ord[i] = 0x007FFFFFu; /* ord(-inf) */ }
    if (i < NNODES * OUT_F) out[i] = 0.f;
}

// ---------------------------------------------------------------------------
// Shared helper: stage 64 gathered/concatenated edge rows into LDS as bf16.
// ---------------------------------------------------------------------------
__device__ __forceinline__ void stage_edge_rows(
        unsigned short* sA, int e0, int wave, int lane,
        const int* __restrict__ ei, const float* __restrict__ x,
        const float* __restrict__ et) {
    for (int j = 0; j < 8; ++j) {
        int r = wave * 8 + j;
        int e = e0 + r;
        int src = ei[e], dst = ei[NEDGES + e];
        const float* xs = x + (long)src * IN_F;
        const float* xd = x + (long)dst * IN_F;
        const float* ep = et + (long)e * NRELA;
        for (int q = 0; q < 10; ++q) {
            int c = lane + q * 32;
            float v = (c < IN_F) ? xs[c]
                    : (c < 2 * IN_F) ? xd[c - IN_F]
                    : ep[c - 2 * IN_F];
            sA[r * A_STRIDE + c] = f2bf_u16(v);
        }
    }
}

// WMMA accumulate one wave's 16x64 strip: mtile rows x 4 n-tiles, K=320.
__device__ __forceinline__ void wmma_strip(
        const unsigned short* sA, const unsigned short* WtU,
        int mtile, int ntbase, int lane, v8f acc[4]) {
    const int arow  = mtile * 16 + (lane & 15);
    const int koffA = (lane >> 4) * 16;     // A frag: lanes16-31 hold K+16..31
    const int koffB = (lane >> 4) * 16;     // B frag: lanes16-31 hold K+16..31
    const __bf16* WtB = (const __bf16*)WtU;
    for (int kt = 0; kt < KTILES; ++kt) {
        const __bf16* ap = (const __bf16*)&sA[arow * A_STRIDE + kt * 32 + koffA];
        v8bf alo = *(const v8bf*)ap;            // K = kb .. kb+7
        v8bf ahi = *(const v8bf*)(ap + 8);      // K = kb+8 .. kb+15
        v16bf af = __builtin_shufflevector(alo, ahi,
                     0,1,2,3,4,5,6,7,8,9,10,11,12,13,14,15);
#pragma unroll
        for (int nt = 0; nt < 4; ++nt) {
            int n = (ntbase + nt) * 16 + (lane & 15);
            const v16bf* bp =
                (const v16bf*)(WtB + (long)n * KDIM + kt * 32 + koffB);
            acc[nt] = __builtin_amdgcn_wmma_f32_16x16x32_bf16(
                          false, af, false, *bp, (short)0, acc[nt],
                          false, false);
        }
    }
}

// ---------------------------------------------------------------------------
// Pass A: edge GEMM via WMMA, alpha = leaky_relu(edge_h @ a),
//         per-node segment max via ordered-uint atomicMax.
// ---------------------------------------------------------------------------
__global__ void __launch_bounds__(256)
gat_edge_alpha(const int* __restrict__ ei, const float* __restrict__ x,
               const float* __restrict__ et, const unsigned short* __restrict__ Wt,
               const float* __restrict__ avec,
               float* __restrict__ alpha, unsigned* __restrict__ m_ord) {
    __shared__ __align__(16) unsigned short sA[ETILE * A_STRIDE];
    __shared__ float alphabuf[ETILE];
    const int tid  = threadIdx.x;
    const int wave = tid >> 5, lane = tid & 31;
    const int e0   = blockIdx.x * ETILE;

    if (tid < ETILE) alphabuf[tid] = 0.f;
    stage_edge_rows(sA, e0, wave, lane, ei, x, et);
    __syncthreads();

    const int mtile  = wave >> 1;
    const int ntbase = (wave & 1) * 4;
    v8f acc[4] = {};
    wmma_strip(sA, Wt, mtile, ntbase, lane, acc);

    // partial dot with 'a' over this wave's 64 n-columns, per owned row
    float pv[8];
#pragma unroll
    for (int v = 0; v < 8; ++v) pv[v] = 0.f;
#pragma unroll
    for (int nt = 0; nt < 4; ++nt) {
        float an = avec[(ntbase + nt) * 16 + (lane & 15)];
#pragma unroll
        for (int v = 0; v < 8; ++v) pv[v] += acc[nt][v] * an;
    }
#pragma unroll
    for (int msk = 1; msk <= 8; msk <<= 1) {
#pragma unroll
        for (int v = 0; v < 8; ++v) pv[v] += __shfl_xor(pv[v], msk, 32);
    }
    if ((lane & 15) == 0) {
        int rbase = mtile * 16 + ((lane >> 4) << 3);
#pragma unroll
        for (int v = 0; v < 8; ++v) atomicAdd(&alphabuf[rbase + v], pv[v]);
    }
    __syncthreads();

    if (tid < ETILE) {
        float ar = alphabuf[tid];
        ar = ar > 0.f ? ar : LRELU * ar;
        int e = e0 + tid;
        alpha[e] = ar;
        atomicMax(&m_ord[ei[e]], f2ord(ar));
    }
}

// ---------------------------------------------------------------------------
// Pass B: e = exp(alpha - m[src]); segment sum via atomicAdd.
// ---------------------------------------------------------------------------
__global__ void __launch_bounds__(256)
gat_expsum(const int* __restrict__ ei, float* __restrict__ alpha,
           const unsigned* __restrict__ m_ord, float* __restrict__ s) {
    int e = blockIdx.x * 256 + threadIdx.x;
    if (e >= NEDGES) return;
    int src = ei[e];
    float ev = __expf(alpha[e] - ord2f(m_ord[src]));
    alpha[e] = ev;
    atomicAdd(&s[src], ev);
}

// ---------------------------------------------------------------------------
// Pass C: recompute edge_h via WMMA, scale rows by alpha/(s+eps),
//         scatter-add into out[src].
// ---------------------------------------------------------------------------
__global__ void __launch_bounds__(256)
gat_aggregate(const int* __restrict__ ei, const float* __restrict__ x,
              const float* __restrict__ et, const unsigned short* __restrict__ Wt,
              const float* __restrict__ alpha, const float* __restrict__ s,
              float* __restrict__ out) {
    __shared__ __align__(16) unsigned short sA[ETILE * A_STRIDE];
    __shared__ float wbuf[ETILE];
    __shared__ int   srcbuf[ETILE];
    const int tid  = threadIdx.x;
    const int wave = tid >> 5, lane = tid & 31;
    const int e0   = blockIdx.x * ETILE;

    if (tid < ETILE) {
        int e = e0 + tid;
        int src = ei[e];
        srcbuf[tid] = src;
        wbuf[tid] = alpha[e] / (s[src] + 1e-16f);
    }
    stage_edge_rows(sA, e0, wave, lane, ei, x, et);
    __syncthreads();

    const int mtile  = wave >> 1;
    const int ntbase = (wave & 1) * 4;
    v8f acc[4] = {};
    wmma_strip(sA, Wt, mtile, ntbase, lane, acc);

    const int half = lane >> 4;
#pragma unroll
    for (int nt = 0; nt < 4; ++nt) {
        int n = (ntbase + nt) * 16 + (lane & 15);
#pragma unroll
        for (int v = 0; v < 8; ++v) {
            int r = mtile * 16 + v + half * 8;
            atomicAdd(&out[(long)srcbuf[r] * OUT_F + n], acc[nt][v] * wbuf[r]);
        }
    }
}

// ---------------------------------------------------------------------------
// Pass D: ELU.
// ---------------------------------------------------------------------------
__global__ void __launch_bounds__(256)
gat_elu(float* __restrict__ out) {
    int i = blockIdx.x * 256 + threadIdx.x;
    if (i < NNODES * OUT_F) {
        float v = out[i];
        out[i] = v > 0.f ? v : __expf(v) - 1.f;
    }
}

// ---------------------------------------------------------------------------
extern "C" void kernel_launch(void* const* d_in, const int* in_sizes, int n_in,
                              void* d_out, int out_size, void* d_ws, size_t ws_size,
                              hipStream_t stream) {
    const int*   ei   = (const int*)d_in[0];     // [2, E]
    const float* x    = (const float*)d_in[1];   // [N, 128]
    const float* et   = (const float*)d_in[2];   // [E, 64]
    const float* W    = (const float*)d_in[3];   // [320, 128]
    const float* avec = (const float*)d_in[4];   // [128, 1]
    float* out = (float*)d_out;

    char* ws = (char*)d_ws;
    unsigned short* Wt    = (unsigned short*)(ws + WS_WT);
    float*          alpha = (float*)(ws + WS_ALPHA);
    float*          s     = (float*)(ws + WS_S);
    unsigned*       m_ord = (unsigned*)(ws + WS_M);

    (void)in_sizes; (void)n_in; (void)out_size; (void)ws_size;

    gat_prep<<<(NNODES * OUT_F) / 256, 256, 0, stream>>>(W, Wt, s, m_ord, out);
    gat_edge_alpha<<<NEDGES / ETILE, 256, 0, stream>>>(ei, x, et, Wt, avec,
                                                       alpha, m_ord);
    gat_expsum<<<NEDGES / 256, 256, 0, stream>>>(ei, alpha, m_ord, s);
    gat_aggregate<<<NEDGES / ETILE, 256, 0, stream>>>(ei, x, et, Wt, alpha, s,
                                                      out);
    gat_elu<<<(NNODES * OUT_F) / 256, 256, 0, stream>>>(out);
}
